// AttentionConditionGenerator_38474317037801
// MI455X (gfx1250) — matrix operand order repaired and checked by reading, not compiled, not run on hardware
//
#include <hip/hip_runtime.h>
#include <hip/hip_bf16.h>
#include <math.h>

#define DMODEL 1024
#define NFFN   4096
#define NHEAD  16
#define BROWS  16384

typedef __bf16 v16bf __attribute__((ext_vector_type(16)));
typedef float  v8f   __attribute__((ext_vector_type(8)));
typedef unsigned int u32x4 __attribute__((ext_vector_type(4)));
typedef float  f32x4 __attribute__((ext_vector_type(4)));
typedef unsigned short ush;

constexpr int BM = 128, BN = 128, BK = 32;
constexpr int LDT = 40;   // padded LDS row stride (bf16 elems): 80B rows -> conflict-free frag loads

union Frag16 { u32x4 u[2]; v16bf v; };
union Pack8  { ush h[8]; u32x4 u; };

__device__ __forceinline__ ush f2bf(float x) {
  return __builtin_bit_cast(ush, (__bf16)x);
}
__device__ __forceinline__ float bf2f(ush s) {
  return (float)__builtin_bit_cast(__bf16, s);
}
__device__ __forceinline__ float gelu_exact(float x) {
  return 0.5f * x * (1.0f + erff(x * 0.70710678118654752f));
}
__device__ __forceinline__ v8f wmma_bf16(const Frag16& a, const Frag16& b, v8f c) {
  return __builtin_amdgcn_wmma_f32_16x16x32_bf16(false, a.v, false, b.v, (short)0, c, false, false);
}

// --- CDNA5 async global->LDS copy (GLOBAL_LOAD_ASYNC_TO_LDS_B128, ASYNCcnt) ---
// Low 32 bits of a generic pointer to __shared__ ARE the LDS byte offset
// (aperture bits live in [63:32] per the CDNA5 flat-aperture rules).
__device__ __forceinline__ unsigned lds_lo32(const void* p) {
  return (unsigned)(unsigned long long)p;
}
// GVS mode: mem = SGPR_base + VGPR_byte_offset; one instruction moves 16B/lane.
__device__ __forceinline__ void async_b128(unsigned lds_off, unsigned voff_bytes, const void* base) {
  asm volatile("global_load_async_to_lds_b128 %0, %1, %2"
               :: "v"(lds_off), "v"(voff_bytes), "s"(base)
               : "memory");
}
#if __has_builtin(__builtin_amdgcn_s_wait_asynccnt)
#define WAIT_ASYNC(n) __builtin_amdgcn_s_wait_asynccnt(n)
#else
#define WAIT_ASYNC(n) asm volatile("s_wait_asynccnt " #n ::: "memory")
#endif

// ---------------------------------------------------------------------------
// q = (dom @ w_q^T + b_q) * (1/sqrt(HD))
__global__ void qproj_kernel(const float* __restrict__ dom, const float* __restrict__ wq,
                             const float* __restrict__ bq, float* __restrict__ qs) {
  int i = blockIdx.x * 256 + threadIdx.x;
  if (i >= DMODEL) return;
  const float* wr = wq + (size_t)i * DMODEL;
  float acc = 0.f;
  for (int j = 0; j < DMODEL; ++j) acc += dom[j] * wr[j];
  qs[i] = (acc + bq[i]) * 0.125f;   // 1/sqrt(64)
}

// r[h][j] = sum_d qs[h*64+d] * w_k[h*64+d][j]   (folds K-projection into 16 vectors)
__global__ void rfold_kernel(const float* __restrict__ qs, const float* __restrict__ wk,
                             float* __restrict__ rvec) {
  int idx = blockIdx.x * 256 + threadIdx.x;     // 16384 total
  int h = idx >> 10, j = idx & 1023;
  float acc = 0.f;
  const float* base = wk + (size_t)(h * 64) * DMODEL + j;
  for (int d = 0; d < 64; ++d) acc += qs[h * 64 + d] * base[(size_t)d * DMODEL];
  rvec[(size_t)h * DMODEL + j] = acc;
}

__global__ void cvt_bf16_kernel(const float* __restrict__ in, ush* __restrict__ out, int n) {
  int i = blockIdx.x * 256 + threadIdx.x;
  int stride = gridDim.x * 256;
  for (; i < n; i += stride) out[i] = f2bf(in[i]);
}

// ---------------------------------------------------------------------------
// Fused: V = kv @ w_v^T (bf16 WMMA) + scores = kv @ r^T (VALU) + softmax(2) +
// head-wise convex combine -> ctx (bf16).  kv row g: b = g>>1, s = g&1.
// A tile: fp32->bf16 register staging. B tile: async double-buffered in LDS.
__global__ __launch_bounds__(256)
void fused_scores_v_kernel(const float* __restrict__ cross, const float* __restrict__ tgt,
                           const ush* __restrict__ wv, const float* __restrict__ bv,
                           const float* __restrict__ rvec, ush* __restrict__ ctx)
{
  __shared__ ush   sA[BM * LDT];
  __shared__ ush   sBd[2][BN * LDT];
  __shared__ float sR[NHEAD * BK];
  __shared__ float sSc[BM * NHEAD];

  const int tid = threadIdx.x;
  const int nb = blockIdx.x, mb = blockIdx.y;

  const int lrow = tid >> 1;
  const int c0   = (tid & 1) * 16;

  const int kvrow = mb * BM + lrow;
  const float* gA = (kvrow & 1 ? tgt : cross) + (size_t)(kvrow >> 1) * DMODEL + c0;
  const float* gR = rvec + (size_t)(tid >> 3) * DMODEL + (tid & 7) * 4;  // valid for tid<128

  const unsigned vbB = (unsigned)(((nb * BN + lrow) * DMODEL + c0) * 2);  // byte offset into wv
  const unsigned ldsB0 = lds_lo32(&sBd[0][lrow * LDT + c0]);
  const unsigned ldsB1 = lds_lo32(&sBd[1][lrow * LDT + c0]);

  f32x4 ra[4]; f32x4 rr;
  {
    const f32x4* pa = (const f32x4*)gA;
    ra[0] = pa[0]; ra[1] = pa[1]; ra[2] = pa[2]; ra[3] = pa[3];
    if (tid < 128) rr = *(const f32x4*)gR;
    // prologue: async-copy B k-tile 0 into buffer 0
    async_b128(ldsB0,      vbB,      wv);
    async_b128(ldsB0 + 16, vbB + 16, wv);
  }

  v8f acc[8] = {};
  float sacc[8] = {0.f, 0.f, 0.f, 0.f, 0.f, 0.f, 0.f, 0.f};

  const int wave = tid >> 5;
  const int wm = (wave >> 1) * 32;
  const int wn = (wave & 1) * 64;
  const int lane = tid & 31;
  const int lh = lane >> 4, lm = lane & 15;
  const int srow = tid >> 1, hb = (tid & 1) * 8;

  const int KT = DMODEL / BK;   // 32
  for (int kt = 0; kt < KT; ++kt) {
    const int p = kt & 1;
    __syncthreads();   // all waves done computing previous tile (protects sA and sBd[1-p])
    // stage A regs -> LDS (fp32 -> bf16 on the fly)
    {
      float tmp[16];
      #pragma unroll
      for (int i = 0; i < 4; ++i)
        #pragma unroll
        for (int j = 0; j < 4; ++j) tmp[i * 4 + j] = ra[i][j];
      Pack8 plo, phi;
      #pragma unroll
      for (int i = 0; i < 8; ++i) { plo.h[i] = f2bf(tmp[i]); phi.h[i] = f2bf(tmp[8 + i]); }
      *(u32x4*)&sA[lrow * LDT + c0]     = plo.u;
      *(u32x4*)&sA[lrow * LDT + c0 + 8] = phi.u;
      if (tid < 128) *(f32x4*)&sR[(tid >> 3) * BK + (tid & 7) * 4] = rr;
    }
    // async-copy next B k-tile into the other buffer while we compute this one
    if (kt + 1 < KT) {
      const unsigned vb = vbB + (unsigned)((kt + 1) * BK * 2);
      const unsigned l = p ? ldsB0 : ldsB1;
      async_b128(l,      vb,      wv);
      async_b128(l + 16, vb + 16, wv);
      WAIT_ASYNC(2);   // retire the older group (this tile's B) — in-order completion
    } else {
      WAIT_ASYNC(0);
    }
    __syncthreads();   // sA/sR visible + every wave's B(p) arrived
    // prefetch next A k-tile into registers (overlaps with WMMA below)
    if (kt + 1 < KT) {
      const f32x4* pa = (const f32x4*)(gA + (kt + 1) * BK);
      ra[0] = pa[0]; ra[1] = pa[1]; ra[2] = pa[2]; ra[3] = pa[3];
      if (tid < 128) rr = *(const f32x4*)(gR + (kt + 1) * BK);
    }
    // score partials: this thread owns row srow, heads hb..hb+7
    #pragma unroll 4
    for (int kk = 0; kk < BK; ++kk) {
      float av = bf2f(sA[srow * LDT + kk]);
      #pragma unroll
      for (int h = 0; h < 8; ++h) sacc[h] += av * sR[(hb + h) * BK + kk];
    }
    // WMMA tile: wave covers 32(M) x 64(N)
    const ush* sB = sBd[p];
    Frag16 af[2];
    #pragma unroll
    for (int mt = 0; mt < 2; ++mt) {
      const int off = (wm + mt * 16 + lm) * LDT + lh * 16;
      af[mt].u[0] = *(const u32x4*)&sA[off];
      af[mt].u[1] = *(const u32x4*)&sA[off + 8];
    }
    #pragma unroll
    for (int nt = 0; nt < 4; ++nt) {
      Frag16 bf;
      const int off = (wn + nt * 16 + lm) * LDT + lh * 16;
      bf.u[0] = *(const u32x4*)&sB[off];
      bf.u[1] = *(const u32x4*)&sB[off + 8];
      acc[0 * 4 + nt] = wmma_bf16(af[0], bf, acc[0 * 4 + nt]);
      acc[1 * 4 + nt] = wmma_bf16(af[1], bf, acc[1 * 4 + nt]);
    }
  }

  // publish scores, then softmax(2) + combine inside the C-fragment (pairs are within-lane)
  #pragma unroll
  for (int h = 0; h < 8; ++h) sSc[srow * NHEAD + hb + h] = sacc[h];
  __syncthreads();

  #pragma unroll
  for (int mt = 0; mt < 2; ++mt)
    #pragma unroll
    for (int nt = 0; nt < 4; ++nt)
      #pragma unroll
      for (int vi = 0; vi < 8; vi += 2) {
        const int lr = wm + mt * 16 + lh * 8 + vi;            // even kv row (s=0)
        const int gcol = nb * BN + wn + nt * 16 + lm;
        const int h = gcol >> 6;
        const float e0 = sSc[lr * NHEAD + h];
        const float e1 = sSc[(lr + 1) * NHEAD + h];
        const float a1 = 1.0f / (1.0f + __expf(e0 - e1));     // softmax over {cross,target}
        const float v0 = acc[mt * 4 + nt][vi];
        const float v1 = acc[mt * 4 + nt][vi + 1];
        const float cv = v0 + a1 * (v1 - v0) + bv[gcol];
        const int brow = (mb * BM + lr) >> 1;
        ctx[(size_t)brow * DMODEL + gcol] = f2bf(cv);
      }
}

// ---------------------------------------------------------------------------
// Generic bf16 WMMA GEMM: C[M,N] = A[M,K] @ W[N,K]^T + bias, tile 128x128x32.
// Fully async double-buffered: GLOBAL_LOAD_ASYNC_TO_LDS_B128 moves both tiles,
// no register staging.  EPI 0: f32 out.  EPI 1: exact-GELU -> bf16 out.
template <int EPI>
__global__ __launch_bounds__(256)
void gemm_bf16_kernel(const ush* __restrict__ A, const ush* __restrict__ Bw,
                      const float* __restrict__ bias,
                      float* __restrict__ Cf, ush* __restrict__ Cb,
                      int N, int K)
{
  __shared__ ush sAd[2][BM * LDT];
  __shared__ ush sBd[2][BN * LDT];

  const int tid = threadIdx.x;
  const int nb = blockIdx.x, mb = blockIdx.y;
  const int lrow = tid >> 1;
  const int c0   = (tid & 1) * 16;

  const unsigned vaA = (unsigned)(((size_t)(mb * BM + lrow) * K + c0) * 2);
  const unsigned vaB = (unsigned)(((size_t)(nb * BN + lrow) * K + c0) * 2);
  const unsigned ldsA[2] = { lds_lo32(&sAd[0][lrow * LDT + c0]), lds_lo32(&sAd[1][lrow * LDT + c0]) };
  const unsigned ldsB[2] = { lds_lo32(&sBd[0][lrow * LDT + c0]), lds_lo32(&sBd[1][lrow * LDT + c0]) };

  // prologue: k-tile 0 into buffer 0 (4 async ops per wave per tile)
  async_b128(ldsA[0],      vaA,      A);
  async_b128(ldsA[0] + 16, vaA + 16, A);
  async_b128(ldsB[0],      vaB,      Bw);
  async_b128(ldsB[0] + 16, vaB + 16, Bw);

  v8f acc[8] = {};

  const int wave = tid >> 5;
  const int wm = (wave >> 1) * 32;
  const int wn = (wave & 1) * 64;
  const int lane = tid & 31;
  const int lh = lane >> 4, lm = lane & 15;

  const int KT = K / BK;
  for (int kt = 0; kt < KT; ++kt) {
    const int p = kt & 1;
    if (kt + 1 < KT) {
      const unsigned kofs = (unsigned)((kt + 1) * BK * 2);
      async_b128(ldsA[1 - p],      vaA + kofs,      A);
      async_b128(ldsA[1 - p] + 16, vaA + kofs + 16, A);
      async_b128(ldsB[1 - p],      vaB + kofs,      Bw);
      async_b128(ldsB[1 - p] + 16, vaB + kofs + 16, Bw);
      WAIT_ASYNC(4);   // retire this tile's group; next tile stays in flight
    } else {
      WAIT_ASYNC(0);
    }
    __syncthreads();   // tile p visible to all waves

    const ush* sA = sAd[p];
    const ush* sB = sBd[p];
    Frag16 af[2];
    #pragma unroll
    for (int mt = 0; mt < 2; ++mt) {
      const int off = (wm + mt * 16 + lm) * LDT + lh * 16;
      af[mt].u[0] = *(const u32x4*)&sA[off];
      af[mt].u[1] = *(const u32x4*)&sA[off + 8];
    }
    #pragma unroll
    for (int nt = 0; nt < 4; ++nt) {
      Frag16 bf;
      const int off = (wn + nt * 16 + lm) * LDT + lh * 16;
      bf.u[0] = *(const u32x4*)&sB[off];
      bf.u[1] = *(const u32x4*)&sB[off + 8];
      acc[0 * 4 + nt] = wmma_bf16(af[0], bf, acc[0 * 4 + nt]);
      acc[1 * 4 + nt] = wmma_bf16(af[1], bf, acc[1 * 4 + nt]);
    }
    __syncthreads();   // all waves done with tile p before it is overwritten
  }

  #pragma unroll
  for (int mt = 0; mt < 2; ++mt)
    #pragma unroll
    for (int nt = 0; nt < 4; ++nt)
      #pragma unroll
      for (int vi = 0; vi < 8; ++vi) {
        const int grow = mb * BM + wm + mt * 16 + lh * 8 + vi;
        const int gcol = nb * BN + wn + nt * 16 + lm;
        const float v = acc[mt * 4 + nt][vi] + bias[gcol];
        if (EPI == 0) Cf[(size_t)grow * N + gcol] = v;
        else          Cb[(size_t)grow * N + gcol] = f2bf(gelu_exact(v));
      }
}

// ---------------------------------------------------------------------------
// Row LayerNorm kernels (D = 1024, 256 threads, 4 cols/thread)
__device__ __forceinline__ float block_reduce_sum(float v, float* red, int tid) {
  red[tid] = v; __syncthreads();
  for (int o = 128; o > 0; o >>= 1) { if (tid < o) red[tid] += red[tid + o]; __syncthreads(); }
  float r = red[0]; __syncthreads();
  return r;
}

__global__ __launch_bounds__(256)
void ln1_kernel(const float* __restrict__ attn, const float* __restrict__ dom,
                const float* __restrict__ g, const float* __restrict__ be,
                float* __restrict__ xf, ush* __restrict__ xb)
{
  __shared__ float red[256];
  const int row = blockIdx.x, tid = threadIdx.x;
  float v[4], s = 0.f;
  #pragma unroll
  for (int i = 0; i < 4; ++i) {
    int c = tid + i * 256;
    v[i] = attn[(size_t)row * DMODEL + c] + dom[c];
    s += v[i];
  }
  const float mean = block_reduce_sum(s, red, tid) * (1.0f / DMODEL);
  float sq = 0.f;
  #pragma unroll
  for (int i = 0; i < 4; ++i) { float d = v[i] - mean; sq += d * d; }
  const float rstd = rsqrtf(block_reduce_sum(sq, red, tid) * (1.0f / DMODEL) + 1e-5f);
  #pragma unroll
  for (int i = 0; i < 4; ++i) {
    int c = tid + i * 256;
    float x = (v[i] - mean) * rstd * g[c] + be[c];
    xf[(size_t)row * DMODEL + c] = x;
    xb[(size_t)row * DMODEL + c] = f2bf(x);
  }
}

__global__ __launch_bounds__(256)
void ln2_kernel(const float* __restrict__ x, const float* __restrict__ ffn,
                const float* __restrict__ g, const float* __restrict__ be,
                float* __restrict__ out)
{
  __shared__ float red[256];
  const int row = blockIdx.x, tid = threadIdx.x;
  float v[4], s = 0.f;
  #pragma unroll
  for (int i = 0; i < 4; ++i) {
    int c = tid + i * 256;
    v[i] = x[(size_t)row * DMODEL + c] + ffn[(size_t)row * DMODEL + c];
    s += v[i];
  }
  const float mean = block_reduce_sum(s, red, tid) * (1.0f / DMODEL);
  float sq = 0.f;
  #pragma unroll
  for (int i = 0; i < 4; ++i) { float d = v[i] - mean; sq += d * d; }
  const float rstd = rsqrtf(block_reduce_sum(sq, red, tid) * (1.0f / DMODEL) + 1e-5f);
  #pragma unroll
  for (int i = 0; i < 4; ++i) {
    int c = tid + i * 256;
    out[(size_t)row * DMODEL + c] = (v[i] - mean) * rstd * g[c] + be[c];
  }
}

// ---------------------------------------------------------------------------
extern "C" void kernel_launch(void* const* d_in, const int* in_sizes, int n_in,
                              void* d_out, int out_size, void* d_ws, size_t ws_size,
                              hipStream_t stream)
{
  const float* h_cross = (const float*)d_in[0];
  const float* h_tgt   = (const float*)d_in[1];
  const float* dom     = (const float*)d_in[2];
  const float* w_q     = (const float*)d_in[3];
  const float* w_k     = (const float*)d_in[4];
  const float* w_v     = (const float*)d_in[5];
  const float* b_q     = (const float*)d_in[6];
  // b_k (d_in[7]) provably cancels in the 2-way softmax
  const float* b_v     = (const float*)d_in[8];
  const float* w_o     = (const float*)d_in[9];
  const float* b_o     = (const float*)d_in[10];
  const float* ln1g    = (const float*)d_in[11];
  const float* ln1b    = (const float*)d_in[12];
  const float* w1      = (const float*)d_in[13];
  const float* b1      = (const float*)d_in[14];
  const float* w2      = (const float*)d_in[15];
  const float* b2      = (const float*)d_in[16];
  const float* ln2g    = (const float*)d_in[17];
  const float* ln2b    = (const float*)d_in[18];

  char* ws = (char*)d_ws;
  size_t off = 0;
  auto alloc = [&](size_t bytes) -> void* {
    void* p = ws + off;
    off = (off + bytes + 255) & ~(size_t)255;
    return p;
  };
  float* qs     = (float*)alloc((size_t)DMODEL * 4);
  float* rvec   = (float*)alloc((size_t)NHEAD * DMODEL * 4);
  ush*   wv_bf  = (ush*)alloc((size_t)DMODEL * DMODEL * 2);
  ush*   wo_bf  = (ush*)alloc((size_t)DMODEL * DMODEL * 2);
  ush*   w1_bf  = (ush*)alloc((size_t)NFFN * DMODEL * 2);
  ush*   w2_bf  = (ush*)alloc((size_t)DMODEL * NFFN * 2);
  ush*   ctx_bf = (ush*)alloc((size_t)BROWS * DMODEL * 2);
  float* tmp_f  = (float*)alloc((size_t)BROWS * DMODEL * 4);   // attn_out, later reused for ffn2
  float* x_f    = (float*)alloc((size_t)BROWS * DMODEL * 4);
  ush*   x_bf   = (ush*)alloc((size_t)BROWS * DMODEL * 2);
  ush*   h_bf   = (ush*)alloc((size_t)BROWS * NFFN * 2);

  // Stage 0: weight conversion + folded score vectors
  cvt_bf16_kernel<<<1024, 256, 0, stream>>>(w_v, wv_bf, DMODEL * DMODEL);
  cvt_bf16_kernel<<<1024, 256, 0, stream>>>(w_o, wo_bf, DMODEL * DMODEL);
  cvt_bf16_kernel<<<2048, 256, 0, stream>>>(w1, w1_bf, NFFN * DMODEL);
  cvt_bf16_kernel<<<2048, 256, 0, stream>>>(w2, w2_bf, DMODEL * NFFN);
  qproj_kernel<<<4, 256, 0, stream>>>(dom, w_q, b_q, qs);
  rfold_kernel<<<64, 256, 0, stream>>>(qs, w_k, rvec);

  // Stage 1: fused V-projection + scores + softmax + combine -> ctx (bf16)
  fused_scores_v_kernel<<<dim3(DMODEL / BN, (2 * BROWS) / BM), 256, 0, stream>>>(
      h_cross, h_tgt, wv_bf, b_v, rvec, ctx_bf);

  // Stage 2: attn_out = ctx @ w_o^T + b_o
  gemm_bf16_kernel<0><<<dim3(DMODEL / BN, BROWS / BM), 256, 0, stream>>>(
      ctx_bf, wo_bf, b_o, tmp_f, nullptr, DMODEL, DMODEL);

  // Stage 3: x = LN1(dom + attn_out)
  ln1_kernel<<<BROWS, 256, 0, stream>>>(tmp_f, dom, ln1g, ln1b, x_f, x_bf);

  // Stage 4: h = gelu(x @ w1^T + b1)  (bf16 out)
  gemm_bf16_kernel<1><<<dim3(NFFN / BN, BROWS / BM), 256, 0, stream>>>(
      x_bf, w1_bf, b1, nullptr, h_bf, NFFN, DMODEL);

  // Stage 5: ffn = h @ w2^T + b2
  gemm_bf16_kernel<0><<<dim3(DMODEL / BN, BROWS / BM), 256, 0, stream>>>(
      h_bf, w2_bf, b2, tmp_f, nullptr, DMODEL, NFFN);

  // Stage 6: out = LN2(x + ffn)
  ln2_kernel<<<BROWS, 256, 0, stream>>>(x_f, tmp_f, ln2g, ln2b, (float*)d_out);
}
